// MultiHeadAttentionBlock_78305843740726
// MI455X (gfx1250) — matrix operand order repaired and checked by reading, not compiled, not run on hardware
//
#include <hip/hip_runtime.h>
#include <hip/hip_bf16.h>

typedef __attribute__((ext_vector_type(16))) _Float16 v16h;
typedef __attribute__((ext_vector_type(8)))  _Float16 v8h;
typedef __attribute__((ext_vector_type(2)))  _Float16 h2;
typedef __attribute__((ext_vector_type(8)))  float    v8f;

#define SHUF16(lo, hi) __builtin_shufflevector(lo, hi, 0,1,2,3,4,5,6,7,8,9,10,11,12,13,14,15)

// ds_swizzle xor-reduce step on packed f16 pair (group-of-32 mode: and=0x1f, xor=M).
#define SWZMAX(p, M)                                                               \
  p = __builtin_elementwise_max(                                                   \
      p, __builtin_bit_cast(h2, __builtin_amdgcn_ds_swizzle(                       \
             __builtin_bit_cast(int, p), ((M) << 10) | 0x1f)))

static __device__ inline v8f wmma_f16(v16h a, v16h b, v8f c) {
  return __builtin_amdgcn_wmma_f32_16x16x32_f16(false, a, false, b, (short)0, c, false, false);
}

// ---------------------------------------------------------------------------
// Elementwise f32 -> f16 conversion (bandwidth bound; done ONCE per tensor so
// the GEMM inner loops carry zero conversion VALU and half the vmem traffic).
// n must be a multiple of 2048 (true for all tensors here).
// ---------------------------------------------------------------------------
__global__ __launch_bounds__(256) void cvt_f16_kernel(const float* __restrict__ src,
                                                      _Float16* __restrict__ dst) {
  const size_t i = ((size_t)blockIdx.x * 256 + threadIdx.x) * 8;
  v8f x = *(const v8f*)(src + i);
  v8h y;
#pragma unroll
  for (int j = 0; j < 8; ++j) y[j] = (_Float16)x[j];
  *(v8h*)(dst + i) = y;
}

// ---------------------------------------------------------------------------
// GEMM: Y[m][n] = sum_k A[m][k] * W[n][k] + bias[n]   (M=4096, N=K=512)
// A, W are f16. One wave -> 16x64 strip (4 WMMA tiles); block = 4 waves.
// MODE : 0 = f16 out, head-split [B,H,S,DK]
//        1 = f16 out, head-split transposed [B,H,DK,S]   (for V^T)
//        2 = f32 out, row-major [4096][512]              (final projection)
// ---------------------------------------------------------------------------
template <int MODE>
__global__ __launch_bounds__(128) void mha_gemm512_kernel(
    const _Float16* __restrict__ A, const _Float16* __restrict__ W,
    const float* __restrict__ bias, void* __restrict__ Out, float scale) {
  constexpr int K = 512;
  const int lane   = threadIdx.x & 31;
  const int wave   = threadIdx.x >> 5;
  const int lane16 = lane & 15;
  const int rhalf  = (lane < 16) ? 0 : 8;
  const int k0a    = (lane < 16) ? 0 : 16;   // A-layout K base for this half-wave
  const int k0b    = (lane < 16) ? 0 : 16;   // B-layout K base for this half-wave
  const int rowbase = blockIdx.y * 64 + wave * 16;
  const int colbase = blockIdx.x * 64;
  const int rowA    = rowbase + lane16;

  v8f acc[4];
#pragma unroll
  for (int t = 0; t < 4; ++t)
#pragma unroll
    for (int r = 0; r < 8; ++r) acc[t][r] = 0.0f;

#pragma unroll 4
  for (int kb = 0; kb < K; kb += 32) {
    // ---- A operand: 16x32 f16 tile in WMMA A layout ----
    const _Float16* ap = A + (size_t)rowA * K + kb + k0a;
    v8h alo = *(const v8h*)ap;
    v8h ahi = *(const v8h*)(ap + 8);
    v16h a = SHUF16(alo, ahi);
    // ---- B operands: 32x16 tiles, B[k][n] = W[n][k] (contiguous 32B) ----
#pragma unroll
    for (int t = 0; t < 4; ++t) {
      const _Float16* wp = W + (size_t)(colbase + t * 16 + lane16) * K + kb + k0b;
      v16h bm = *(const v16h*)wp;
      acc[t] = wmma_f16(a, bm, acc[t]);
    }
  }

  // ---- epilogue: bias, scale, scatter per output mode ----
#pragma unroll
  for (int t = 0; t < 4; ++t) {
    const int col = colbase + t * 16 + lane16;
    const float bv = bias[col];
#pragma unroll
    for (int r = 0; r < 8; ++r) {
      const int grow = rowbase + r + rhalf;           // 0..4095
      float val = (acc[t][r] + bv) * scale;
      if constexpr (MODE == 2) {
        ((float*)Out)[(size_t)grow * 512 + col] = val;
      } else {
        const int b = grow >> 11, s = grow & 2047;    // batch, seq
        const int h = col >> 6,  dk = col & 63;       // head, head-dim
        _Float16* o = (_Float16*)Out;
        if constexpr (MODE == 0)
          o[(((size_t)(b * 8 + h) * 2048) + s) * 64 + dk] = (_Float16)val;
        else
          o[(((size_t)(b * 8 + h) * 64) + dk) * 2048 + s] = (_Float16)val;
      }
    }
  }
}

// ---------------------------------------------------------------------------
// Causal attention, flash-style online softmax (log2 domain).
// One wave (32 threads) per block; each block owns one (b, h, 16-row q tile).
// Qh/Kh: f16 [B,H,S,64] (Q pre-scaled by log2(e)/sqrt(64)); Vt: f16 [B,H,64,S].
// Row sums are produced by an extra WMMA against an all-ones B matrix; the only
// cross-lane reduction is the f16-packed row max via ds_swizzle. The key loop
// is split: (qrow0>>5) mask-free tiles + exactly one masked diagonal tile.
// Output: f16 row-major [4096][512] (heads re-merged) for the final GEMM.
// ---------------------------------------------------------------------------
__global__ __launch_bounds__(32) void mha_attn_kernel(
    const _Float16* __restrict__ Qh, const _Float16* __restrict__ Kh,
    const _Float16* __restrict__ Vt, _Float16* __restrict__ AT) {
  __shared__ _Float16 lp[16 * 32];  // P-tile bounce buffer (C-layout -> A-layout)

  const int lane   = threadIdx.x & 31;
  const int lane16 = lane & 15;
  const int rhalf  = (lane < 16) ? 0 : 8;
  const int k0a    = (lane < 16) ? 0 : 16;
  const int k0b    = (lane < 16) ? 0 : 16;

  const int qt    = blockIdx.x;       // 0..2047
  const int bh    = qt >> 7;          // (b*8 + h)
  const int qi    = qt & 127;
  const int qrow0 = qi << 4;
  const int b     = bh >> 3;
  const int h     = bh & 7;

  const _Float16* Qp  = Qh + (size_t)bh * 2048 * 64;
  const _Float16* Kp  = Kh + (size_t)bh * 2048 * 64;
  const _Float16* Vtp = Vt + (size_t)bh * 64 * 2048;

  // Load Q tile (16x64) as two A-layout operands (K-chunks of 32).
  v16h aq[2];
  {
    const int arow = qrow0 + lane16;
#pragma unroll
    for (int c = 0; c < 2; ++c) {
      const _Float16* ap = Qp + (size_t)arow * 64 + c * 32 + k0a;
      v8h lo = *(const v8h*)ap;
      v8h hi = *(const v8h*)(ap + 8);
      aq[c] = SHUF16(lo, hi);
    }
  }

  v16h ones;
#pragma unroll
  for (int i = 0; i < 16; ++i) ones[i] = (_Float16)1.0f;

  v8f o[4], lacc;
#pragma unroll
  for (int t = 0; t < 4; ++t)
#pragma unroll
    for (int r = 0; r < 8; ++r) o[t][r] = 0.0f;
#pragma unroll
  for (int r = 0; r < 8; ++r) lacc[r] = 0.0f;
  float mrow[8];
#pragma unroll
  for (int r = 0; r < 8; ++r) mrow[r] = -1e30f;

  auto tile = [&](int kbase, bool masked) __attribute__((always_inline)) {
    // Prefetch next K tile (gfx1250 global_prefetch_b8).
    __builtin_prefetch(Kp + (size_t)(kbase + 32) * 64, 0, 1);

    // ---- scores: two 16x16 tiles over 32 keys (log2 domain) ----
    v8f s0, s1;
#pragma unroll
    for (int r = 0; r < 8; ++r) { s0[r] = 0.0f; s1[r] = 0.0f; }
#pragma unroll
    for (int c = 0; c < 2; ++c) {
      const _Float16* kp0 = Kp + (size_t)(kbase + lane16) * 64 + c * 32 + k0b;
      const _Float16* kp1 = Kp + (size_t)(kbase + 16 + lane16) * 64 + c * 32 + k0b;
      v16h bk0 = *(const v16h*)kp0;
      v16h bk1 = *(const v16h*)kp1;
      s0 = wmma_f16(aq[c], bk0, s0);
      s1 = wmma_f16(aq[c], bk1, s1);
    }

    // ---- causal mask: only the single diagonal tile pays for it ----
    if (masked) {
      const int col0 = kbase + lane16;
      const int col1 = col0 + 16;
#pragma unroll
      for (int r = 0; r < 8; ++r) {
        const int row = qrow0 + r + rhalf;
        s0[r] = (col0 <= row) ? s0[r] : -1e30f;
        s1[r] = (col1 <= row) ? s1[r] : -1e30f;
      }
    }

    // ---- row max: pack row pairs to f16, xor-reduce across 16 lanes ----
    h2 pk[4];
#pragma unroll
    for (int j = 0; j < 4; ++j) {
      pk[j][0] = (_Float16)fmaxf(s0[2 * j],     s1[2 * j]);
      pk[j][1] = (_Float16)fmaxf(s0[2 * j + 1], s1[2 * j + 1]);
    }
#pragma unroll
    for (int j = 0; j < 4; ++j) {
      SWZMAX(pk[j], 1);
      SWZMAX(pk[j], 2);
      SWZMAX(pk[j], 4);
      SWZMAX(pk[j], 8);
    }

    // ---- online softmax update (exact for any shift sequence) ----
    float corr[8];
#pragma unroll
    for (int r = 0; r < 8; ++r) {
      float tmax = (float)pk[r >> 1][r & 1];
      float mnew = fmaxf(mrow[r], tmax);
      corr[r] = __builtin_amdgcn_exp2f(mrow[r] - mnew);
      mrow[r] = mnew;
      s0[r] = __builtin_amdgcn_exp2f(s0[r] - mnew);
      s1[r] = __builtin_amdgcn_exp2f(s1[r] - mnew);
    }
#pragma unroll
    for (int t = 0; t < 4; ++t)
#pragma unroll
      for (int r = 0; r < 8; ++r) o[t][r] *= corr[r];
#pragma unroll
    for (int r = 0; r < 8; ++r) lacc[r] *= corr[r];

    // ---- P: C-layout -> A-layout via LDS ----
#pragma unroll
    for (int r = 0; r < 8; ++r) {
      const int row = r + rhalf;
      lp[row * 32 + lane16]      = (_Float16)s0[r];
      lp[row * 32 + 16 + lane16] = (_Float16)s1[r];
    }
    __syncthreads();
    v8h plo = *(const v8h*)(lp + lane16 * 32 + k0a);
    v8h phi = *(const v8h*)(lp + lane16 * 32 + k0a + 8);
    v16h pa = SHUF16(plo, phi);
    __syncthreads();

    // ---- row sums on the matrix unit: l += P @ ones ----
    lacc = wmma_f16(pa, ones, lacc);

    // ---- O += P @ V  (V^T rows are contiguous over keys) ----
#pragma unroll
    for (int t = 0; t < 4; ++t) {
      const _Float16* vp = Vtp + (size_t)(t * 16 + lane16) * 2048 + kbase + k0b;
      v16h bv = *(const v16h*)vp;
      o[t] = wmma_f16(pa, bv, o[t]);
    }
  };

  const int nfull = qrow0 >> 5;            // fully-unmasked 32-key tiles
  for (int kt = 0; kt < nfull; ++kt) tile(kt * 32, false);
  tile(nfull * 32, true);                  // the one diagonal tile

  // ---- normalize and write merged-head f16 output [4096][512] ----
#pragma unroll
  for (int t = 0; t < 4; ++t) {
#pragma unroll
    for (int r = 0; r < 8; ++r) {
      const int row = qrow0 + r + rhalf;
      const int gr  = b * 2048 + row;
      const int gc  = h * 64 + t * 16 + lane16;
      AT[(size_t)gr * 512 + gc] = (_Float16)(o[t][r] / lacc[r]);
    }
  }
}

extern "C" void kernel_launch(void* const* d_in, const int* in_sizes, int n_in,
                              void* d_out, int out_size, void* d_ws, size_t ws_size,
                              hipStream_t stream) {
  (void)in_sizes; (void)n_in; (void)out_size; (void)ws_size;
  const float* q  = (const float*)d_in[0];
  const float* k  = (const float*)d_in[1];
  const float* v  = (const float*)d_in[2];
  // d_in[3] is the causal mask; causality is hard-coded in the attn kernel.
  const float* Wq = (const float*)d_in[4];
  const float* bq = (const float*)d_in[5];
  const float* Wk = (const float*)d_in[6];
  const float* bk = (const float*)d_in[7];
  const float* Wv = (const float*)d_in[8];
  const float* bv = (const float*)d_in[9];
  const float* Wo = (const float*)d_in[10];
  const float* bo = (const float*)d_in[11];

  // Workspace (f16 elements): 3 x 2M (X) + 4 x 256K (W) + 4 x 2M (Q/K/Vt/AT) = 30 MB.
  constexpr size_t NX = 4096 * 512;   // 2,097,152
  constexpr size_t NW = 512 * 512;    //   262,144
  _Float16* Xq  = (_Float16*)d_ws;
  _Float16* Xk  = Xq + NX;
  _Float16* Xv  = Xk + NX;
  _Float16* Wqh = Xv + NX;
  _Float16* Wkh = Wqh + NW;
  _Float16* Wvh = Wkh + NW;
  _Float16* Woh = Wvh + NW;
  _Float16* Qh  = Woh + NW;           // [B,H,S,DK], scaled log2(e)/8
  _Float16* Kh  = Qh + NX;            // [B,H,S,DK]
  _Float16* Vt  = Kh + NX;            // [B,H,DK,S]
  _Float16* AT  = Vt + NX;            // [4096][512] merged heads

  // One-time f32 -> f16 conversions (removes all cvt work from GEMM loops).
  cvt_f16_kernel<<<dim3(NX / 2048), dim3(256), 0, stream>>>(q, Xq);
  cvt_f16_kernel<<<dim3(NX / 2048), dim3(256), 0, stream>>>(k, Xk);
  cvt_f16_kernel<<<dim3(NX / 2048), dim3(256), 0, stream>>>(v, Xv);
  cvt_f16_kernel<<<dim3(NW / 2048), dim3(256), 0, stream>>>(Wq, Wqh);
  cvt_f16_kernel<<<dim3(NW / 2048), dim3(256), 0, stream>>>(Wk, Wkh);
  cvt_f16_kernel<<<dim3(NW / 2048), dim3(256), 0, stream>>>(Wv, Wvh);
  cvt_f16_kernel<<<dim3(NW / 2048), dim3(256), 0, stream>>>(Wo, Woh);

  dim3 gg(8, 64, 1);   // 512/64 col tiles, 4096/64 row tiles
  dim3 gb(128, 1, 1);  // 4 waves

  // Fold softmax scale AND log2(e) into Q so scores live in the log2 domain.
  const float qscale = 0.125f * 1.44269504088896340736f;

  mha_gemm512_kernel<0><<<gg, gb, 0, stream>>>(Xq, Wqh, bq, Qh, qscale);
  mha_gemm512_kernel<0><<<gg, gb, 0, stream>>>(Xk, Wkh, bk, Kh, 1.0f);
  mha_gemm512_kernel<1><<<gg, gb, 0, stream>>>(Xv, Wvh, bv, Vt, 1.0f);

  mha_attn_kernel<<<dim3(2048), dim3(32), 0, stream>>>(Qh, Kh, Vt, AT);

  mha_gemm512_kernel<2><<<gg, gb, 0, stream>>>(AT, Woh, bo, d_out, 1.0f);
}
